// KGATLayer_59090160058522
// MI455X (gfx1250) — compile-verified, matrix operand order checked by reading, not run on hardware
//
#include <hip/hip_runtime.h>
#include <hip/hip_bf16.h>
#include <math.h>

// ---------------------------------------------------------------------------
// KGAT layer, factored form.
//   PS = ent_embed @ (W_ent@W2a)            (100000x128, WMMA f16 GEMM)
//   PD = ent_embed @ (W_ent@W2b)            (fused in same GEMM pass)
//   PR[r] = (rel_embed@W_rel + b_rel)@W2c + b_ent@(W2a+W2b) + b_ent2  (237x128)
//   s_e  = qS[src] + qD[dst] + qR[rel] + a_b   (q* = dot(P*, a_w))
//   b_e  = exp(leaky_relu(s_e))
//   h[n] = PS[n] + (1/sum b_e) * sum b_e*(PD[dst]+PR[rel])   (since sum alpha = 1)
// ---------------------------------------------------------------------------

typedef __attribute__((ext_vector_type(16))) _Float16 v16h;
typedef __attribute__((ext_vector_type(8)))  _Float16 v8h;
typedef __attribute__((ext_vector_type(8)))  float    v8f;

#define N_NODES 100000
#define N_RELS  237
#define N_EDGES 500000
#define D 128

// ---------------- workspace layout (bytes) ----------------
#define WS_MAT   (0)                         // MaT f16 128x128 (N-major)
#define WS_MBT   (WS_MAT + 32768)            // MbT f16 128x128 (N-major)
#define WS_RM    (WS_MBT + 32768)            // Rm  f32 237x128
#define WS_CB    (WS_RM + 121600)            // cb  f32 128
#define WS_PR    (WS_CB + 512)               // PR' f32 237x128
#define WS_QR    (WS_PR + 121600)            // qR  f32 237
#define WS_QS    (WS_QR + 1024)              // qS  f32 100000
#define WS_QD    (WS_QS + 400128)            // qD  f32 100000
#define WS_BS    (WS_QD + 400128)            // bs  f32 100000
#define WS_PS    (WS_BS + 400128)            // PS  f32 100000x128
#define WS_PD    (WS_PS + 51200000)          // PD  f32 100000x128

__global__ __launch_bounds__(256) void k_zero(float* p, int n) {
    int i = blockIdx.x * 256 + threadIdx.x;
    if (i < n) p[i] = 0.0f;
}

// MaT[n*128+i] = (f16) sum_j W_ent[i][j] * W_ent2[j][n]        (block a)
// MbT[n*128+i] = (f16) sum_j W_ent[i][j] * W_ent2[128+j][n]    (block b)
__global__ __launch_bounds__(256) void k_weights(const float* __restrict__ W_ent,
                                                 const float* __restrict__ W2,
                                                 _Float16* __restrict__ MaT,
                                                 _Float16* __restrict__ MbT) {
    int idx = blockIdx.x * 256 + threadIdx.x;   // 128*128 threads
    int n = idx >> 7, i = idx & 127;
    float sa = 0.f, sb = 0.f;
    #pragma unroll 4
    for (int j = 0; j < D; ++j) {
        float w = W_ent[i * D + j];
        sa = fmaf(w, W2[j * D + n], sa);
        sb = fmaf(w, W2[(D + j) * D + n], sb);
    }
    MaT[n * D + i] = (_Float16)sa;
    MbT[n * D + i] = (_Float16)sb;
}

// cb[n] = sum_j b_ent[j]*(W2a[j][n]+W2b[j][n]) + b_ent2[n]
__global__ __launch_bounds__(128) void k_cbias(const float* __restrict__ b_ent,
                                               const float* __restrict__ W2,
                                               const float* __restrict__ b_ent2,
                                               float* __restrict__ cb) {
    int n = threadIdx.x;
    float s = b_ent2[n];
    for (int j = 0; j < D; ++j)
        s = fmaf(b_ent[j], W2[j * D + n] + W2[(D + j) * D + n], s);
    cb[n] = s;
}

// Rm[r][j] = sum_i rel_embed[r][i]*W_rel[i][j] + b_rel[j]
__global__ __launch_bounds__(256) void k_rel1(const float* __restrict__ rel_embed,
                                              const float* __restrict__ W_rel,
                                              const float* __restrict__ b_rel,
                                              float* __restrict__ Rm) {
    int idx = blockIdx.x * 256 + threadIdx.x;
    if (idx >= N_RELS * D) return;
    int r = idx >> 7, j = idx & 127;
    float s = b_rel[j];
    for (int i = 0; i < D; ++i)
        s = fmaf(rel_embed[r * D + i], W_rel[i * D + j], s);
    Rm[idx] = s;
}

// PR'[r][n] = sum_j Rm[r][j]*W2c[j][n] + cb[n]
__global__ __launch_bounds__(256) void k_rel2(const float* __restrict__ Rm,
                                              const float* __restrict__ W2,
                                              const float* __restrict__ cb,
                                              float* __restrict__ PR) {
    int idx = blockIdx.x * 256 + threadIdx.x;
    if (idx >= N_RELS * D) return;
    int r = idx >> 7, n = idx & 127;
    float s = cb[n];
    for (int j = 0; j < D; ++j)
        s = fmaf(Rm[r * D + j], W2[(2 * D + j) * D + n], s);
    PR[idx] = s;
}

// ---------------------------------------------------------------------------
// Fused WMMA GEMM:  PS = X@Ma, PD = X@Mb.  X = ent_embed (f32->f16 in LDS).
// Workgroup = 256 threads = 8 waves; block of 32 rows, full N=128 for both
// weight matrices.  Wave w: row-tile (w&1), col group (w>>1 -> cols 32*(w>>1)),
// 4 accumulators (2 col tiles x {Ma,Mb}), K loop 128/32 -> 16 WMMAs/wave.
// A/B 16-bit fragment layout (ISA 7.12.2): lane l (l<16) holds M(or N)=l,
// halves [kb..kb+7] and [kb+16..kb+23]; lane l>=16 holds [kb+8..15],[kb+24..31].
// ---------------------------------------------------------------------------
#define XS_STRIDE 136   // 128 + 8 halves pad -> conflict-free row access

__global__ __launch_bounds__(256) void k_gemm(const float* __restrict__ X,
                                              const _Float16* __restrict__ MaT,
                                              const _Float16* __restrict__ MbT,
                                              float* __restrict__ PS,
                                              float* __restrict__ PD) {
    __shared__ _Float16 Xs[32 * XS_STRIDE];
    const int tid = threadIdx.x;
    const int rowBase = blockIdx.x * 32;

    // Stage 32x128 f32 tile -> f16 LDS. Thread t: row t/8, cols (t%8)*16..+15.
    {
        const int r = tid >> 3;
        const int c0 = (tid & 7) * 16;
        const float4* X4 = (const float4*)X;
        _Float16* dst = &Xs[r * XS_STRIDE + c0];
        #pragma unroll
        for (int j = 0; j < 4; ++j) {
            float4 f = X4[(size_t)(rowBase + r) * 32 + (c0 >> 2) + j];
            dst[j * 4 + 0] = (_Float16)f.x;
            dst[j * 4 + 1] = (_Float16)f.y;
            dst[j * 4 + 2] = (_Float16)f.z;
            dst[j * 4 + 3] = (_Float16)f.w;
        }
    }
    __syncthreads();

    const int wave = tid >> 5;
    const int lane = tid & 31;
    const int rt = wave & 1;          // row tile 0/1
    const int cg = wave >> 1;         // col group 0..3 -> cols [32*cg, 32*cg+32)
    const int lmod = lane & 15;
    const int lhi8 = (lane >> 4) << 3;   // 0 or 8

    v8f c0a = {}, c1a = {}, c0b = {}, c1b = {};

    const int arow = rt * 16 + lmod;
    const int n0 = cg * 32 + lmod;       // col tile 0
    const int n1 = n0 + 16;              // col tile 1

    #pragma unroll
    for (int kk = 0; kk < 4; ++kk) {
        const int kb = kk * 32 + lhi8;
        // A fragment from LDS
        v8h alo = *(const v8h*)&Xs[arow * XS_STRIDE + kb];
        v8h ahi = *(const v8h*)&Xs[arow * XS_STRIDE + kb + 16];
        v16h a;
        #pragma unroll
        for (int i = 0; i < 8; ++i) { a[i] = alo[i]; a[8 + i] = ahi[i]; }

        // B fragments from N-major weights (rows L2-resident)
        v16h b0a, b1a, b0b, b1b;
        {
            v8h lo = *(const v8h*)&MaT[n0 * D + kb];
            v8h hi = *(const v8h*)&MaT[n0 * D + kb + 16];
            #pragma unroll
            for (int i = 0; i < 8; ++i) { b0a[i] = lo[i]; b0a[8 + i] = hi[i]; }
        }
        {
            v8h lo = *(const v8h*)&MaT[n1 * D + kb];
            v8h hi = *(const v8h*)&MaT[n1 * D + kb + 16];
            #pragma unroll
            for (int i = 0; i < 8; ++i) { b1a[i] = lo[i]; b1a[8 + i] = hi[i]; }
        }
        {
            v8h lo = *(const v8h*)&MbT[n0 * D + kb];
            v8h hi = *(const v8h*)&MbT[n0 * D + kb + 16];
            #pragma unroll
            for (int i = 0; i < 8; ++i) { b0b[i] = lo[i]; b0b[8 + i] = hi[i]; }
        }
        {
            v8h lo = *(const v8h*)&MbT[n1 * D + kb];
            v8h hi = *(const v8h*)&MbT[n1 * D + kb + 16];
            #pragma unroll
            for (int i = 0; i < 8; ++i) { b1b[i] = lo[i]; b1b[8 + i] = hi[i]; }
        }

        c0a = __builtin_amdgcn_wmma_f32_16x16x32_f16(false, a, false, b0a, (short)0, c0a, false, false);
        c1a = __builtin_amdgcn_wmma_f32_16x16x32_f16(false, a, false, b1a, (short)0, c1a, false, false);
        c0b = __builtin_amdgcn_wmma_f32_16x16x32_f16(false, a, false, b0b, (short)0, c0b, false, false);
        c1b = __builtin_amdgcn_wmma_f32_16x16x32_f16(false, a, false, b1b, (short)0, c1b, false, false);
    }

    // Epilogue: C layout (ISA 7.12.2): VGPR i, lanes0-15 -> M=i, lanes16-31 -> M=8+i; N=lane%16.
    const int mBase = rowBase + rt * 16 + ((lane >> 4) << 3);
    const int col0 = cg * 32 + lmod;
    #pragma unroll
    for (int i = 0; i < 8; ++i) {
        size_t row = (size_t)(mBase + i) * D;
        PS[row + col0]      = c0a[i];
        PS[row + col0 + 16] = c1a[i];
        PD[row + col0]      = c0b[i];
        PD[row + col0 + 16] = c1b[i];
    }
}

// q[row] = dot(M[row], a_w) ; one wave32 per row
__global__ __launch_bounds__(256) void k_dot(const float* __restrict__ M,
                                             const float* __restrict__ aw,
                                             float* __restrict__ q, int rows) {
    int wave = (blockIdx.x * 256 + threadIdx.x) >> 5;
    int lane = threadIdx.x & 31;
    if (wave >= rows) return;
    const float* m = M + (size_t)wave * D + lane * 4;
    float s = m[0] * aw[lane * 4] + m[1] * aw[lane * 4 + 1] +
              m[2] * aw[lane * 4 + 2] + m[3] * aw[lane * 4 + 3];
    #pragma unroll
    for (int off = 16; off; off >>= 1) s += __shfl_down(s, off, 32);
    if (lane == 0) q[wave] = s;
}

// Per-edge: b_e = exp(leaky_relu(qS+qD+qR+a_b)); acc[src]+= b_e*(PD[dst]+PR[rel]);
// bs[src]+= b_e.  128 threads per edge (one per channel), 2 edges per block.
__global__ __launch_bounds__(256) void k_edge(const int* __restrict__ trip,
                                              const float* __restrict__ qS,
                                              const float* __restrict__ qD,
                                              const float* __restrict__ qR,
                                              const float* __restrict__ ab,
                                              const float* __restrict__ PD,
                                              const float* __restrict__ PR,
                                              float* __restrict__ acc,
                                              float* __restrict__ bs) {
    int e = blockIdx.x * 2 + (threadIdx.x >> 7);
    int ch = threadIdx.x & 127;
    int s = trip[3 * e], d = trip[3 * e + 1], r = trip[3 * e + 2];
    float sc = qS[s] + qD[d] + qR[r] + ab[0];
    float lr = sc > 0.f ? sc : 0.01f * sc;
    float b = __expf(lr);
    float val = b * (PD[(size_t)d * D + ch] + PR[r * D + ch]);
    unsafeAtomicAdd(&acc[(size_t)s * D + ch], val);
    if (ch == 0) unsafeAtomicAdd(&bs[s], b);
}

// h[n][c] = bs[n]!=0 ? PS[n][c] + acc[n][c]/bs[n] : 0   (in place in d_out)
__global__ __launch_bounds__(256) void k_final(const float* __restrict__ PS,
                                               const float* __restrict__ bs,
                                               float* __restrict__ out) {
    int idx = blockIdx.x * 256 + threadIdx.x;
    int n = idx >> 7;
    float denom = bs[n];
    out[idx] = (denom != 0.f) ? fmaf(out[idx], __frcp_rn(denom), PS[idx]) : 0.f;
}

extern "C" void kernel_launch(void* const* d_in, const int* in_sizes, int n_in,
                              void* d_out, int out_size, void* d_ws, size_t ws_size,
                              hipStream_t stream) {
    const int*   trip      = (const int*)d_in[0];
    const float* ent_embed = (const float*)d_in[1];
    const float* rel_embed = (const float*)d_in[2];
    const float* W_ent     = (const float*)d_in[3];
    const float* b_ent     = (const float*)d_in[4];
    const float* W_rel     = (const float*)d_in[5];
    const float* b_rel     = (const float*)d_in[6];
    const float* W_ent2    = (const float*)d_in[7];
    const float* b_ent2    = (const float*)d_in[8];
    const float* a_w       = (const float*)d_in[9];
    const float* a_b       = (const float*)d_in[10];
    float* out = (float*)d_out;
    char* ws = (char*)d_ws;

    _Float16* MaT = (_Float16*)(ws + WS_MAT);
    _Float16* MbT = (_Float16*)(ws + WS_MBT);
    float* Rm = (float*)(ws + WS_RM);
    float* cb = (float*)(ws + WS_CB);
    float* PR = (float*)(ws + WS_PR);
    float* qR = (float*)(ws + WS_QR);
    float* qS = (float*)(ws + WS_QS);
    float* qD = (float*)(ws + WS_QD);
    float* bs = (float*)(ws + WS_BS);
    float* PS = (float*)(ws + WS_PS);
    float* PD = (float*)(ws + WS_PD);

    // zero accumulators (harness poisons buffers; must re-zero every call)
    k_zero<<<(N_NODES * D + 255) / 256, 256, 0, stream>>>(out, N_NODES * D);
    k_zero<<<(N_NODES + 255) / 256, 256, 0, stream>>>(bs, N_NODES);

    // fold weight chains
    k_weights<<<(D * D) / 256, 256, 0, stream>>>(W_ent, W_ent2, MaT, MbT);
    k_cbias<<<1, 128, 0, stream>>>(b_ent, W_ent2, b_ent2, cb);
    k_rel1<<<(N_RELS * D + 255) / 256, 256, 0, stream>>>(rel_embed, W_rel, b_rel, Rm);
    k_rel2<<<(N_RELS * D + 255) / 256, 256, 0, stream>>>(Rm, W_ent2, cb, PR);

    // node GEMMs via WMMA (100000 rows, 32 per workgroup)
    k_gemm<<<N_NODES / 32, 256, 0, stream>>>(ent_embed, MaT, MbT, PS, PD);

    // attention scalar components
    k_dot<<<(N_NODES + 7) / 8, 256, 0, stream>>>(PS, a_w, qS, N_NODES);
    k_dot<<<(N_NODES + 7) / 8, 256, 0, stream>>>(PD, a_w, qD, N_NODES);
    k_dot<<<(N_RELS + 7) / 8, 256, 0, stream>>>(PR, a_w, qR, N_RELS);

    // edge scatter (accumulate into d_out) and finalize
    k_edge<<<N_EDGES / 2, 256, 0, stream>>>(trip, qS, qD, qR, a_b, PD, PR, out, bs);
    k_final<<<(N_NODES * D) / 256, 256, 0, stream>>>(PS, bs, out);
}
